// MultiHeadSelfAttention_33767032881328
// MI455X (gfx1250) — matrix operand order repaired and checked by reading, not compiled, not run on hardware
//
#include <hip/hip_runtime.h>

// MI455X (gfx1250) mixed-precision MHA:
//   f32 -> bf16 once, all matmuls via v_wmma_f32_16x16x32_bf16 (f32 accum),
//   RoPE in registers, flash-attention with online softmax, f32 output.
//   Tile staging uses GLOBAL_LOAD_ASYNC_TO_LDS_B128 (ASYNCcnt) with
//   double-buffered LDS in the GEMMs for copy/compute overlap.
//   sched_group_barrier pins the ds_read/wmma interleave so accumulators
//   stay stationary (no v_mov_b64 shuffles / hazard v_nops in hot loops).
// Workspace layout (bf16 elements): Xb | Wqkvb | Woutb | Qb | Kb | Vb | Ob  (~201 MB)

typedef unsigned short u16;
typedef unsigned int   u32;
typedef unsigned long long u64;
typedef __attribute__((ext_vector_type(16))) __bf16 bf16x16;
typedef __attribute__((ext_vector_type(8)))  float  floatx8;

#define D_MODEL 2048
#define N_HEADS 16
#define D_HEAD  128
#define SEQ     2048
#define BATCH   4
#define BS      (BATCH * SEQ)   // 8192 rows
#define NQKV    (3 * D_MODEL)   // 6144

union Frag  { uint4 u[2]; bf16x16 v; };  // 32 bytes = one 16x32 bf16 wmma operand per lane
union Pack8 { u16 s[8]; uint4 u; };

static __device__ __forceinline__ u16 f2bf(float f) {
  union { float f; u32 u; } x; x.f = f;
  u32 r = x.u + 0x7fffu + ((x.u >> 16) & 1u);   // round-to-nearest-even
  return (u16)(r >> 16);
}
static __device__ __forceinline__ floatx8 zero8() {
  floatx8 z;
#pragma unroll
  for (int i = 0; i < 8; ++i) z[i] = 0.0f;
  return z;
}
static __device__ __forceinline__ float halfmax(float v) {
#pragma unroll
  for (int m = 8; m >= 1; m >>= 1) v = fmaxf(v, __shfl_xor(v, m, 16));
  return v;
}
static __device__ __forceinline__ float halfsum(float v) {
#pragma unroll
  for (int m = 8; m >= 1; m >>= 1) v += __shfl_xor(v, m, 16);
  return v;
}
static __device__ __forceinline__ floatx8 wmma_bf16(const Frag& a, const Frag& b, floatx8 c) {
  return __builtin_amdgcn_wmma_f32_16x16x32_bf16(false, a.v, false, b.v, (short)0, c, false, false);
}

// ---- CDNA5 async global->LDS copy (ASYNCcnt), GV addressing mode ----
static __device__ __forceinline__ void async_g2l_b128(const u16* g, const u16* l) {
  asm volatile("global_load_async_to_lds_b128 %0, %1, off"
               :: "v"((u32)(unsigned long long)(uintptr_t)l),
                  "v"((u64)(uintptr_t)g)
               : "memory");
}
static __device__ __forceinline__ void wait_async_le4() {
  asm volatile("s_wait_asynccnt 0x4" ::: "memory");
}
static __device__ __forceinline__ void wait_async0() {
  asm volatile("s_wait_asynccnt 0x0" ::: "memory");
}

// ---- scheduling pipeline for one 18-ds_read / 8-wmma tile step ----
// 0x100 = DS read, 0x008 = MFMA/WMMA.  Keeps ~2 B-frags in flight ahead of
// each WMMA while capping live fragments (accumulators stay in place).
static __device__ __forceinline__ void sched_tile() {
  __builtin_amdgcn_sched_group_barrier(0x100, 6, 0);   // A frag + 2 B frags
#pragma unroll
  for (int i = 0; i < 6; ++i) {
    __builtin_amdgcn_sched_group_barrier(0x008, 1, 0); // 1 wmma
    __builtin_amdgcn_sched_group_barrier(0x100, 2, 0); // next B frag
  }
  __builtin_amdgcn_sched_group_barrier(0x008, 1, 0);   // drain
  __builtin_amdgcn_sched_group_barrier(0x008, 1, 0);
}

// ---------------------------------------------------------------- conversion
__global__ __launch_bounds__(256) void k_f32_to_bf16(const float* __restrict__ src,
                                                     u16* __restrict__ dst, int n8) {
  int i = blockIdx.x * 256 + threadIdx.x;
  if (i >= n8) return;
  const float4* s4 = (const float4*)src;
  float4 a = s4[2 * i], b = s4[2 * i + 1];
  Pack8 p;
  p.s[0] = f2bf(a.x); p.s[1] = f2bf(a.y); p.s[2] = f2bf(a.z); p.s[3] = f2bf(a.w);
  p.s[4] = f2bf(b.x); p.s[5] = f2bf(b.y); p.s[6] = f2bf(b.z); p.s[7] = f2bf(b.w);
  ((uint4*)dst)[i] = p.u;
}

// ---------------------------------------------------- QKV GEMM + RoPE epilogue
__global__ __launch_bounds__(256) void k_qkv_rope(
    const u16* __restrict__ Xb, const u16* __restrict__ Wq,
    const int* __restrict__ pos_ids,
    u16* __restrict__ Qb, u16* __restrict__ Kb, u16* __restrict__ Vb) {
  const int tid = threadIdx.x;
  const int w = tid >> 5, lane = tid & 31, l15 = lane & 15, hl = lane >> 4;
  const int n0 = blockIdx.x * 128;
  const int m0 = blockIdx.y * 128;
  __shared__ u16 As[2][128 * 32];
  __shared__ u16 Bs[2][128 * 32];
  floatx8 acc[8];
#pragma unroll
  for (int i = 0; i < 8; ++i) acc[i] = zero8();

  const int r0 = tid >> 2, c0 = (tid & 3) << 3;   // per-thread tile coords
#pragma unroll
  for (int i = 0; i < 2; ++i) {
    int r = r0 + i * 64;
    async_g2l_b128(&Xb[(size_t)(m0 + r) * D_MODEL + c0], &As[0][r * 32 + c0]);
    async_g2l_b128(&Wq[(size_t)(n0 + r) * D_MODEL + c0], &Bs[0][r * 32 + c0]);
  }
  for (int kt = 0; kt < D_MODEL / 32; ++kt) {
    const int cur = kt & 1;
    if (kt + 1 < D_MODEL / 32) {
      const int k1 = (kt + 1) * 32;
#pragma unroll
      for (int i = 0; i < 2; ++i) {
        int r = r0 + i * 64;
        async_g2l_b128(&Xb[(size_t)(m0 + r) * D_MODEL + k1 + c0], &As[cur ^ 1][r * 32 + c0]);
        async_g2l_b128(&Wq[(size_t)(n0 + r) * D_MODEL + k1 + c0], &Bs[cur ^ 1][r * 32 + c0]);
      }
      wait_async_le4();   // in-order completion => tile kt has landed
    } else {
      wait_async0();
    }
    __syncthreads();
    Frag fa;
    const u16* ar = &As[cur][(w * 16 + l15) * 32];
    fa.u[0] = *(const uint4*)(ar + hl * 8);
    fa.u[1] = *(const uint4*)(ar + 16 + hl * 8);
#pragma unroll
    for (int nt = 0; nt < 8; ++nt) {
      Frag fb;
      const u16* br = &Bs[cur][(nt * 16 + l15) * 32];
      fb.u[0] = *(const uint4*)(br + hl * 8);
      fb.u[1] = *(const uint4*)(br + 16 + hl * 8);
      acc[nt] = wmma_bf16(fa, fb, acc[nt]);
    }
    sched_tile();
    __syncthreads();      // buf[cur] free before next iteration writes it
  }

  const int which = n0 >> 11;                 // 0=Q, 1=K, 2=V
  const int h = (n0 & (D_MODEL - 1)) >> 7;    // head index
#pragma unroll
  for (int r = 0; r < 8; ++r) {
    int row = m0 + w * 16 + r + hl * 8;       // C layout: VGPR r -> rows r / r+8
    int b = row >> 11, s = row & (SEQ - 1);
    size_t base = ((size_t)(b * N_HEADS + h) * SEQ + s) * D_HEAD;
    if (which == 2) {
#pragma unroll
      for (int nt = 0; nt < 8; ++nt) Vb[base + nt * 16 + l15] = f2bf(acc[nt][r]);
    } else {
      u16* dst = (which == 0) ? Qb : Kb;
      float pos = (float)pos_ids[s];
#pragma unroll
      for (int nt = 0; nt < 4; ++nt) {
        int d = nt * 16 + l15;                          // 0..63
        float invf = exp2f(-0.2076205059f * (float)d);  // theta^(-2d/128)
        float ang = pos * invf;
        float sn, cs;
        sincosf(ang, &sn, &cs);
        float t1 = acc[nt][r], t2 = acc[nt + 4][r];     // pair (d, d+64)
        dst[base + d]      = f2bf(t1 * cs - t2 * sn);
        dst[base + d + 64] = f2bf(t2 * cs + t1 * sn);
      }
    }
  }
}

// ------------------------------------------------------------- flash attention
__global__ __launch_bounds__(128) void k_attn(
    const u16* __restrict__ Qb, const u16* __restrict__ Kb,
    const u16* __restrict__ Vb, u16* __restrict__ Ob) {
  const int tid = threadIdx.x;
  const int w = tid >> 5, lane = tid & 31, l15 = lane & 15, hl = lane >> 4;
  const int q0 = blockIdx.x * 64;
  const int bh = blockIdx.y;
  const int b = bh >> 4, h = bh & (N_HEADS - 1);
  const size_t head = (size_t)bh * SEQ * D_HEAD;

  __shared__ u16 Qs[64 * 128];        // Q tile       (16 KB)
  __shared__ u16 KV[128 * 128];       // K then V^T   (32 KB, reused per phase)
  __shared__ u16 Ps[4][16 * 128];     // P per wave   (16 KB)

  // Q tile: async global->LDS (waited before the first compute barrier)
#pragma unroll
  for (int i = 0; i < 8; ++i) {
    int idx = tid + i * 128;
    int r = idx >> 4, c = (idx & 15) << 3;
    async_g2l_b128(&Qb[head + (size_t)(q0 + r) * D_HEAD + c], &Qs[r * 128 + c]);
  }
  floatx8 oacc[8];
  float mrow[8], lrow[8];
#pragma unroll
  for (int i = 0; i < 8; ++i) { oacc[i] = zero8(); mrow[i] = -3.0e38f; lrow[i] = 0.0f; }

  const int nkb = (q0 + 63) / 128 + 1;       // causal: only key blocks <= q tile
  const float scale = 0.08838834764831845f;  // 1/sqrt(128)

  for (int kb = 0; kb < nkb; ++kb) {
    const int kbase = kb * 128;
    __syncthreads();                         // KV free (prev PV reads done)
#pragma unroll
    for (int i = 0; i < 16; ++i) {
      int idx = tid + i * 128;
      int r = idx >> 4, c = (idx & 15) << 3;
      async_g2l_b128(&Kb[head + (size_t)(kbase + r) * D_HEAD + c], &KV[r * 128 + c]);
    }
    wait_async0();
    __syncthreads();

    // S = Q K^T : 8 key sub-tiles x 4 K-dim steps = 32 WMMAs
    floatx8 st[8];
#pragma unroll
    for (int i = 0; i < 8; ++i) st[i] = zero8();
#pragma unroll
    for (int kk = 0; kk < 128; kk += 32) {
      Frag fa;
      const u16* ar = &Qs[(w * 16 + l15) * 128 + kk];
      fa.u[0] = *(const uint4*)(ar + hl * 8);
      fa.u[1] = *(const uint4*)(ar + 16 + hl * 8);
#pragma unroll
      for (int nt = 0; nt < 8; ++nt) {
        Frag fb;
        const u16* br = &KV[(nt * 16 + l15) * 128 + kk];
        fb.u[0] = *(const uint4*)(br + hl * 8);
        fb.u[1] = *(const uint4*)(br + 16 + hl * 8);
        st[nt] = wmma_bf16(fa, fb, st[nt]);
      }
      sched_tile();
    }
    // scale + causal mask
#pragma unroll
    for (int nt = 0; nt < 8; ++nt) {
      int key = kbase + nt * 16 + l15;
#pragma unroll
      for (int r = 0; r < 8; ++r) {
        float v = st[nt][r] * scale;
        if (key > q0 + w * 16 + hl * 8 + r) v = -1.0e30f;
        st[nt][r] = v;
      }
    }
    // online softmax (row stats per (r, lane-half) row)
#pragma unroll
    for (int r = 0; r < 8; ++r) {
      float rm = -3.0e38f;
#pragma unroll
      for (int nt = 0; nt < 8; ++nt) rm = fmaxf(rm, st[nt][r]);
      rm = halfmax(rm);
      float mnew = fmaxf(mrow[r], rm);
      float alpha = __expf(mrow[r] - mnew);
      mrow[r] = mnew;
      float psum = 0.0f;
#pragma unroll
      for (int nt = 0; nt < 8; ++nt) {
        float p = __expf(st[nt][r] - mnew);
        st[nt][r] = p;
        psum += p;
      }
      lrow[r] = lrow[r] * alpha + halfsum(psum);
#pragma unroll
      for (int dt = 0; dt < 8; ++dt) oacc[dt][r] *= alpha;
      int lr = r + hl * 8;
#pragma unroll
      for (int nt = 0; nt < 8; ++nt) Ps[w][lr * 128 + nt * 16 + l15] = f2bf(st[nt][r]);
    }
    __syncthreads();
    // V block transposed into KV: KV[d][key] so P*V B-frags are contiguous loads
#pragma unroll
    for (int i = 0; i < 16; ++i) {
      int idx = tid + i * 128;
      int key = idx >> 4, dc = (idx & 15) << 3;
      uint4 vv = *(const uint4*)&Vb[head + (size_t)(kbase + key) * D_HEAD + dc];
      const u16* pv = (const u16*)&vv;
#pragma unroll
      for (int j = 0; j < 8; ++j) KV[(dc + j) * 128 + key] = pv[j];
    }
    __syncthreads();
    // O += P * V : 4 key steps x 8 d sub-tiles = 32 WMMAs
#pragma unroll
    for (int kk = 0; kk < 128; kk += 32) {
      Frag fa;
      const u16* ar = &Ps[w][l15 * 128 + kk];
      fa.u[0] = *(const uint4*)(ar + hl * 8);
      fa.u[1] = *(const uint4*)(ar + 16 + hl * 8);
#pragma unroll
      for (int dt = 0; dt < 8; ++dt) {
        Frag fb;
        const u16* br = &KV[(dt * 16 + l15) * 128 + kk];
        fb.u[0] = *(const uint4*)(br + hl * 8);
        fb.u[1] = *(const uint4*)(br + 16 + hl * 8);
        oacc[dt] = wmma_bf16(fa, fb, oacc[dt]);
      }
      sched_tile();
    }
  }
  // normalize + store O in [b, s, h*dh] layout (ready for out-projection)
#pragma unroll
  for (int r = 0; r < 8; ++r) {
    float inv = 1.0f / lrow[r];
    int srow = q0 + w * 16 + r + hl * 8;
    size_t base = ((size_t)b * SEQ + srow) * D_MODEL + h * D_HEAD;
#pragma unroll
    for (int dt = 0; dt < 8; ++dt) Ob[base + dt * 16 + l15] = f2bf(oacc[dt][r] * inv);
  }
}

// ------------------------------------------------------------ output GEMM (f32)
__global__ __launch_bounds__(256) void k_outproj(
    const u16* __restrict__ Ob, const u16* __restrict__ Wb, float* __restrict__ out) {
  const int tid = threadIdx.x;
  const int w = tid >> 5, lane = tid & 31, l15 = lane & 15, hl = lane >> 4;
  const int n0 = blockIdx.x * 128;
  const int m0 = blockIdx.y * 128;
  __shared__ u16 As[2][128 * 32];
  __shared__ u16 Bs[2][128 * 32];
  floatx8 acc[8];
#pragma unroll
  for (int i = 0; i < 8; ++i) acc[i] = zero8();

  const int r0 = tid >> 2, c0 = (tid & 3) << 3;
#pragma unroll
  for (int i = 0; i < 2; ++i) {
    int r = r0 + i * 64;
    async_g2l_b128(&Ob[(size_t)(m0 + r) * D_MODEL + c0], &As[0][r * 32 + c0]);
    async_g2l_b128(&Wb[(size_t)(n0 + r) * D_MODEL + c0], &Bs[0][r * 32 + c0]);
  }
  for (int kt = 0; kt < D_MODEL / 32; ++kt) {
    const int cur = kt & 1;
    if (kt + 1 < D_MODEL / 32) {
      const int k1 = (kt + 1) * 32;
#pragma unroll
      for (int i = 0; i < 2; ++i) {
        int r = r0 + i * 64;
        async_g2l_b128(&Ob[(size_t)(m0 + r) * D_MODEL + k1 + c0], &As[cur ^ 1][r * 32 + c0]);
        async_g2l_b128(&Wb[(size_t)(n0 + r) * D_MODEL + k1 + c0], &Bs[cur ^ 1][r * 32 + c0]);
      }
      wait_async_le4();
    } else {
      wait_async0();
    }
    __syncthreads();
    Frag fa;
    const u16* ar = &As[cur][(w * 16 + l15) * 32];
    fa.u[0] = *(const uint4*)(ar + hl * 8);
    fa.u[1] = *(const uint4*)(ar + 16 + hl * 8);
#pragma unroll
    for (int nt = 0; nt < 8; ++nt) {
      Frag fb;
      const u16* br = &Bs[cur][(nt * 16 + l15) * 32];
      fb.u[0] = *(const uint4*)(br + hl * 8);
      fb.u[1] = *(const uint4*)(br + 16 + hl * 8);
      acc[nt] = wmma_bf16(fa, fb, acc[nt]);
    }
    sched_tile();
    __syncthreads();
  }
#pragma unroll
  for (int r = 0; r < 8; ++r) {
    int row = m0 + w * 16 + r + hl * 8;
#pragma unroll
    for (int nt = 0; nt < 8; ++nt)
      out[(size_t)row * D_MODEL + n0 + nt * 16 + l15] = acc[nt][r];
  }
}

// ------------------------------------------------------------------- launcher
extern "C" void kernel_launch(void* const* d_in, const int* in_sizes, int n_in,
                              void* d_out, int out_size, void* d_ws, size_t ws_size,
                              hipStream_t stream) {
  (void)in_sizes; (void)n_in; (void)out_size; (void)ws_size;
  const float* x    = (const float*)d_in[0];
  const int*   pos  = (const int*)d_in[1];
  const float* Wqkv = (const float*)d_in[2];
  const float* Wout = (const float*)d_in[3];

  u16* ws  = (u16*)d_ws;
  u16* Xb  = ws;
  u16* Wqb = Xb  + (size_t)BS * D_MODEL;       // 16.7M elems
  u16* Wob = Wqb + (size_t)NQKV * D_MODEL;     // 12.6M
  u16* Qb  = Wob + (size_t)D_MODEL * D_MODEL;  // 4.2M
  u16* Kb  = Qb  + (size_t)BS * D_MODEL;
  u16* Vb  = Kb  + (size_t)BS * D_MODEL;
  u16* Ob  = Vb  + (size_t)BS * D_MODEL;

  int nx = BS * D_MODEL / 8, nq = NQKV * D_MODEL / 8, nw = D_MODEL * D_MODEL / 8;
  k_f32_to_bf16<<<(nx + 255) / 256, 256, 0, stream>>>(x, Xb, nx);
  k_f32_to_bf16<<<(nq + 255) / 256, 256, 0, stream>>>(Wqkv, Wqb, nq);
  k_f32_to_bf16<<<(nw + 255) / 256, 256, 0, stream>>>(Wout, Wob, nw);

  k_qkv_rope<<<dim3(NQKV / 128, BS / 128), 256, 0, stream>>>(Xb, Wqb, pos, Qb, Kb, Vb);
  k_attn<<<dim3(SEQ / 64, BATCH * N_HEADS), 128, 0, stream>>>(Qb, Kb, Vb, Ob);
  k_outproj<<<dim3(D_MODEL / 128, BS / 128), 256, 0, stream>>>(Ob, Wob, (float*)d_out);
}
